// BEVProjection_24524263260793
// MI455X (gfx1250) — matrix operand order repaired and checked by reading, not compiled, not run on hardware
//
#include <hip/hip_runtime.h>

typedef __attribute__((ext_vector_type(4)))  _Float16 v4h;
typedef __attribute__((ext_vector_type(8)))  _Float16 v8h;
typedef __attribute__((ext_vector_type(16))) _Float16 v16h;
typedef __attribute__((ext_vector_type(8)))  float    v8f;

#define CAT16(lo,hi) __builtin_shufflevector(lo, hi, 0,1,2,3,4,5,6,7,8,9,10,11,12,13,14,15)

#define CIN   128
#define COUT  64
#define HH    256
#define WW    256
#define NPTS  100000
#define BB    4
#define KPAD  160      // 131 padded to 5 k-tiles of 32
#define KCONV 576      // 9 taps * 64 ci

// workspace byte offsets (all 256-aligned)
#define OFF_W1T   0u            // 64*160 f16  = 20480 B
#define OFF_W2T   20480u        // 64*64  f16  =  8192 B
#define OFF_CW1T  28672u        // 64*576 f16  = 73728 B
#define OFF_CW2T  102400u       // 64*576 f16  = 73728 B
#define OFF_B1    176128u
#define OFF_B2    176384u
#define OFF_CB1   176640u
#define OFF_CB2   176896u
#define OFF_BEV   262144u       // 4*256*256*64 f32 = 67108864 B (NHWC)
#define OFF_MID   67371008u     // 4*256*256*64 f16 = 33554432 B (NHWC)

// ---------------------------------------------------------------------------
// Prep: fold BN into weights/biases, transpose weights to [out][K] f16
// ---------------------------------------------------------------------------
__global__ __launch_bounds__(256) void prep_kernel(
    const float* __restrict__ w1,  const float* __restrict__ b1,
    const float* __restrict__ g1,  const float* __restrict__ be1,
    const float* __restrict__ m1,  const float* __restrict__ v1,
    const float* __restrict__ w2,  const float* __restrict__ b2,
    const float* __restrict__ cw1, const float* __restrict__ cb1,
    const float* __restrict__ cg1, const float* __restrict__ cbe1,
    const float* __restrict__ cm1, const float* __restrict__ cv1,
    const float* __restrict__ cw2, const float* __restrict__ cb2,
    const float* __restrict__ cg2, const float* __restrict__ cbe2,
    const float* __restrict__ cm2, const float* __restrict__ cv2,
    _Float16* __restrict__ w1t, _Float16* __restrict__ w2t,
    _Float16* __restrict__ cw1t, _Float16* __restrict__ cw2t,
    float* __restrict__ b1f, float* __restrict__ b2f,
    float* __restrict__ cb1f, float* __restrict__ cb2f)
{
    const int tid = blockIdx.x * blockDim.x + threadIdx.x;
    const int nth = gridDim.x * blockDim.x;

    for (int e = tid; e < COUT * KPAD; e += nth) {
        int o = e / KPAD, c = e % KPAD;
        float s = g1[o] * rsqrtf(v1[o] + 1e-5f);
        float v = (c < CIN + 3) ? w1[c * COUT + o] * s : 0.f;
        w1t[e] = (_Float16)v;
    }
    for (int e = tid; e < COUT * COUT; e += nth) {
        int o = e / COUT, c = e % COUT;
        w2t[e] = (_Float16)w2[c * COUT + o];
    }
    for (int e = tid; e < COUT * KCONV; e += nth) {
        int o = e / KCONV, k = e % KCONV;
        int t = k / COUT, i = k % COUT, ky = t / 3, kx = t % 3;
        float s1 = cg1[o] * rsqrtf(cv1[o] + 1e-5f);
        float s2 = cg2[o] * rsqrtf(cv2[o] + 1e-5f);
        cw1t[e] = (_Float16)(cw1[((o * COUT + i) * 3 + ky) * 3 + kx] * s1);
        cw2t[e] = (_Float16)(cw2[((o * COUT + i) * 3 + ky) * 3 + kx] * s2);
    }
    if (tid < COUT) {
        float s = g1[tid] * rsqrtf(v1[tid] + 1e-5f);
        b1f[tid] = b1[tid] * s + be1[tid] - m1[tid] * s;
        b2f[tid] = b2[tid];
        float s1 = cg1[tid] * rsqrtf(cv1[tid] + 1e-5f);
        cb1f[tid] = cb1[tid] * s1 + cbe1[tid] - cm1[tid] * s1;
        float s2 = cg2[tid] * rsqrtf(cv2[tid] + 1e-5f);
        cb2f[tid] = cb2[tid] * s2 + cbe2[tid] - cm2[tid] * s2;
    }
}

// ---------------------------------------------------------------------------
// Point MLP (WMMA f16) + scatter-max into NHWC BEV grid.
// 2 waves / block, each wave owns a 16-point tile. 25000 tiles exactly.
// ---------------------------------------------------------------------------
__global__ __launch_bounds__(64) void mlp_scatter_kernel(
    const float* __restrict__ points, const float* __restrict__ feats,
    const _Float16* __restrict__ w1t, const float* __restrict__ b1f,
    const _Float16* __restrict__ w2t, const float* __restrict__ b2f,
    float* __restrict__ bev)
{
    __shared__ alignas(16) _Float16 sW1[COUT * KPAD];   // 20480 B
    __shared__ alignas(16) _Float16 sW2[COUT * COUT];   //  8192 B
    __shared__ alignas(16) _Float16 sA[2][16 * KPAD];   // 10240 B
    __shared__ alignas(16) _Float16 sH[2][16 * COUT];   //  4096 B
    __shared__ int sMeta[2][16][2];

    const int tid = threadIdx.x;

    { // weights -> LDS
        const uint4* g1v = (const uint4*)w1t;  uint4* l1v = (uint4*)sW1;
        for (int i = tid; i < COUT * KPAD / 8; i += 64) l1v[i] = g1v[i];
        const uint4* g2v = (const uint4*)w2t;  uint4* l2v = (uint4*)sW2;
        for (int i = tid; i < COUT * COUT / 8; i += 64) l2v[i] = g2v[i];
    }

    const int wave = tid >> 5, lane = tid & 31;
    const long tb = ((long)blockIdx.x * 2 + wave) * 16;

    float posx = 0.f, posy = 0.f, posz = 0.f;
    if (lane < 16) {
        long g = tb + lane;
        float px = points[g * 3 + 0], py = points[g * 3 + 1], pz = points[g * 3 + 2];
        int valid = (px >= -50.f) & (px < 50.f) & (py >= -50.f) & (py < 50.f) &
                    (pz >= -3.f) & (pz < 5.f);
        int col = (int)((px + 50.f) * 2.56f);  col = col < 0 ? 0 : (col > 255 ? 255 : col);
        int row = (int)((py + 50.f) * 2.56f);  row = row < 0 ? 0 : (row > 255 ? 255 : row);
        int b = (int)(g / NPTS);
        sMeta[wave][lane][0] = valid;
        sMeta[wave][lane][1] = (((b * HH + row) * WW) + col) * COUT;
        posx = (px + 50.f) * 0.01f;
        posy = (py + 50.f) * 0.01f;
        posz = (pz + 3.f) * 0.125f;
    }

    // stage A tile: 16 points x 160 (features f32 -> f16, pos, zero pad)
    for (int r = 0; r < 16; ++r) {
        long g = tb + r;
        float4 f = *(const float4*)(feats + g * CIN + lane * 4);
        v4h h4; h4[0] = (_Float16)f.x; h4[1] = (_Float16)f.y;
        h4[2] = (_Float16)f.z; h4[3] = (_Float16)f.w;
        *(v4h*)&sA[wave][r * KPAD + lane * 4] = h4;
    }
    if (lane < 16) {
        int r = lane;
        sA[wave][r * KPAD + 128] = (_Float16)posx;
        sA[wave][r * KPAD + 129] = (_Float16)posy;
        sA[wave][r * KPAD + 130] = (_Float16)posz;
        for (int c = 131; c < KPAD; ++c) sA[wave][r * KPAD + c] = (_Float16)0.f;
    }
    __syncthreads();

    const int rA   = lane & 15;
    const int selA = (lane < 16) ? 0 : 8;
    const int selB = (lane < 16) ? 0 : 8;
    const int nCol = lane & 15;

    // ---- GEMM1: (16 x 160) x (160 x 64), fused BN bias + ReLU
    v8f acc[4];
    #pragma unroll
    for (int nt = 0; nt < 4; ++nt) {
        float bv = b1f[nt * 16 + nCol];
        #pragma unroll
        for (int i = 0; i < 8; ++i) acc[nt][i] = bv;
    }
    #pragma unroll
    for (int kt = 0; kt < 5; ++kt) {
        int ka = kt * 32 + selA;
        v8h alo = *(const v8h*)&sA[wave][rA * KPAD + ka];
        v8h ahi = *(const v8h*)&sA[wave][rA * KPAD + ka + 16];
        v16h a = CAT16(alo, ahi);
        int kb = kt * 32 + selB;
        #pragma unroll
        for (int nt = 0; nt < 4; ++nt) {
            int N = nt * 16 + nCol;
            v8h blo = *(const v8h*)&sW1[N * KPAD + kb];
            v8h bhi = *(const v8h*)&sW1[N * KPAD + kb + 16];
            v16h bb = CAT16(blo, bhi);
            acc[nt] = __builtin_amdgcn_wmma_f32_16x16x32_f16(
                false, a, false, bb, (short)0, acc[nt], false, false);
        }
    }
    #pragma unroll
    for (int nt = 0; nt < 4; ++nt)
        #pragma unroll
        for (int i = 0; i < 8; ++i) {
            float v = acc[nt][i]; v = v > 0.f ? v : 0.f;
            int M = (lane < 16) ? i : i + 8;
            sH[wave][M * COUT + nt * 16 + nCol] = (_Float16)v;
        }
    __syncthreads();

    // ---- GEMM2: (16 x 64) x (64 x 64) + b2
    v8f acc2[4];
    #pragma unroll
    for (int nt = 0; nt < 4; ++nt) {
        float bv = b2f[nt * 16 + nCol];
        #pragma unroll
        for (int i = 0; i < 8; ++i) acc2[nt][i] = bv;
    }
    #pragma unroll
    for (int kt = 0; kt < 2; ++kt) {
        int ka = kt * 32 + selA;
        v8h alo = *(const v8h*)&sH[wave][rA * COUT + ka];
        v8h ahi = *(const v8h*)&sH[wave][rA * COUT + ka + 16];
        v16h a = CAT16(alo, ahi);
        int kb = kt * 32 + selB;
        #pragma unroll
        for (int nt = 0; nt < 4; ++nt) {
            int N = nt * 16 + nCol;
            v8h blo = *(const v8h*)&sW2[N * COUT + kb];
            v8h bhi = *(const v8h*)&sW2[N * COUT + kb + 16];
            v16h bb = CAT16(blo, bhi);
            acc2[nt] = __builtin_amdgcn_wmma_f32_16x16x32_f16(
                false, a, false, bb, (short)0, acc2[nt], false, false);
        }
    }

    // ---- scatter-max: int atomicMax on non-negative float bit patterns
    #pragma unroll
    for (int nt = 0; nt < 4; ++nt) {
        int ch = nt * 16 + nCol;
        #pragma unroll
        for (int i = 0; i < 8; ++i) {
            int M = (lane < 16) ? i : i + 8;
            float v = acc2[nt][i];
            if (v > 0.f && sMeta[wave][M][0]) {
                atomicMax((int*)bev + sMeta[wave][M][1] + ch, __float_as_int(v));
            }
        }
    }
}

// ---------------------------------------------------------------------------
// 3x3 conv, 64->64 ch, implicit GEMM via WMMA. 8 waves/block, each wave = 16
// output pixels x 64 channels. Block = one (b, h, 128-pixel segment).
// All 9 taps of weights LDS-resident (72 KB) + 48.75 KB halo tile: uses the
// large-LDS capability of the gfx1250 WGP (320 KB), 2 blocks/WGP.
// FIRST=1: in f32 NHWC bev -> out f16 NHWC (BN+ReLU folded)
// FIRST=0: in f16 NHWC mid -> out f32 NCHW (BN+ReLU folded)
// ---------------------------------------------------------------------------
template<int FIRST>
__global__ __launch_bounds__(256) void conv3x3_kernel(
    const void* __restrict__ in_, const _Float16* __restrict__ wT,
    const float* __restrict__ bias, void* __restrict__ out_)
{
    __shared__ alignas(16) _Float16 sIn[3 * 130 * COUT];  // 49920 B, reused for store staging
    __shared__ alignas(16) _Float16 sWt[COUT * KCONV];    // 73728 B, all 9 taps

    const int tid  = threadIdx.x;
    const int seg  = blockIdx.x & 1;
    const int h    = (blockIdx.x >> 1) & (HH - 1);
    const int b    = blockIdx.x >> 9;
    const int cbase = seg * 128;

    const float*    inF = (const float*)in_;
    const _Float16* inH = (const _Float16*)in_;

    // all conv weights -> LDS (4608 v8h copies, 18 per thread)
    for (int i = tid; i < COUT * KCONV / 8; i += 256) {
        *(v8h*)&sWt[i * 8] = *(const v8h*)&wT[i * 8];
    }

    // load 3 x 130 x 64 halo tile (zero padded at borders)
    for (int e = tid; e < 3 * 130 * 8; e += 256) {
        int r = e / (130 * 8); int rem = e - r * 130 * 8;
        int c = rem >> 3;      int q = rem & 7;
        int gh = h + r - 1;    int gw = cbase + c - 1;
        v8h vals;
        #pragma unroll
        for (int i = 0; i < 8; ++i) vals[i] = (_Float16)0.f;
        if ((unsigned)gh < (unsigned)HH && (unsigned)gw < (unsigned)WW) {
            long base = ((long)(b * HH + gh) * WW + gw) * COUT + q * 8;
            if (FIRST) {
                float4 f0 = *(const float4*)(inF + base);
                float4 f1 = *(const float4*)(inF + base + 4);
                vals[0] = (_Float16)f0.x; vals[1] = (_Float16)f0.y;
                vals[2] = (_Float16)f0.z; vals[3] = (_Float16)f0.w;
                vals[4] = (_Float16)f1.x; vals[5] = (_Float16)f1.y;
                vals[6] = (_Float16)f1.z; vals[7] = (_Float16)f1.w;
            } else {
                vals = *(const v8h*)(inH + base);
            }
        }
        *(v8h*)&sIn[(r * 130 + c) * COUT + q * 8] = vals;
    }
    __syncthreads();

    const int wave = tid >> 5, lane = tid & 31;
    const int rA   = lane & 15;
    const int selA = (lane < 16) ? 0 : 8;
    const int selB = (lane < 16) ? 0 : 8;
    const int nCol = lane & 15;

    v8f acc[4];
    #pragma unroll
    for (int nt = 0; nt < 4; ++nt) {
        float bv = bias[nt * 16 + nCol];
        #pragma unroll
        for (int i = 0; i < 8; ++i) acc[nt][i] = bv;
    }

    #pragma unroll
    for (int t = 0; t < 9; ++t) {
        const int dy = t / 3, dx = t % 3;
        const int colIdx = wave * 16 + rA + dx;      // 0..129 in halo coords
        const _Float16* arow = &sIn[(dy * 130 + colIdx) * COUT];
        #pragma unroll
        for (int sub = 0; sub < 2; ++sub) {
            int ka = sub * 32 + selA;
            v8h alo = *(const v8h*)(arow + ka);
            v8h ahi = *(const v8h*)(arow + ka + 16);
            v16h a = CAT16(alo, ahi);
            int kb = sub * 32 + selB;
            #pragma unroll
            for (int nt = 0; nt < 4; ++nt) {
                int N = nt * 16 + nCol;
                v8h blo = *(const v8h*)&sWt[N * KCONV + t * COUT + kb];
                v8h bhi = *(const v8h*)&sWt[N * KCONV + t * COUT + kb + 16];
                v16h bb = CAT16(blo, bhi);
                acc[nt] = __builtin_amdgcn_wmma_f32_16x16x32_f16(
                    false, a, false, bb, (short)0, acc[nt], false, false);
            }
        }
    }
    __syncthreads();       // done with sIn; reuse as store staging

    if (FIRST) {
        _Float16* stage = sIn;   // 8 waves * 16*64 f16 = 8192 halves <= 24960
        #pragma unroll
        for (int nt = 0; nt < 4; ++nt)
            #pragma unroll
            for (int i = 0; i < 8; ++i) {
                float v = acc[nt][i]; v = v > 0.f ? v : 0.f;
                int M = (lane < 16) ? i : i + 8;
                stage[wave * 1024 + M * COUT + nt * 16 + nCol] = (_Float16)v;
            }
        __syncthreads();
        _Float16* outH = (_Float16*)out_;
        int pix = lane & 15, half = lane >> 4;
        long gb = ((long)(b * HH + h) * WW + cbase + wave * 16 + pix) * COUT + half * 32;
        const uint4* s4 = (const uint4*)&stage[wave * 1024 + pix * COUT + half * 32];
        uint4* o4 = (uint4*)(outH + gb);
        o4[0] = s4[0]; o4[1] = s4[1]; o4[2] = s4[2]; o4[3] = s4[3];
    } else {
        float* stage = (float*)sIn;  // 8 waves * 1024 f32 = 32768 B <= 49920
        #pragma unroll
        for (int nt = 0; nt < 4; ++nt)
            #pragma unroll
            for (int i = 0; i < 8; ++i) {
                float v = acc[nt][i]; v = v > 0.f ? v : 0.f;
                int M = (lane < 16) ? i : i + 8;
                stage[wave * 1024 + (nt * 16 + nCol) * 16 + M] = v;   // [ch][pix]
            }
        __syncthreads();
        float* outF = (float*)out_;
        #pragma unroll
        for (int cc = 0; cc < 2; ++cc) {
            int ch = lane * 2 + cc;
            long gb = ((long)(b * COUT + ch) * HH + h) * WW + cbase + wave * 16;
            const float4* s4 = (const float4*)&stage[wave * 1024 + ch * 16];
            float4* o4 = (float4*)(outF + gb);
            o4[0] = s4[0]; o4[1] = s4[1]; o4[2] = s4[2]; o4[3] = s4[3];
        }
    }
}

// ---------------------------------------------------------------------------
extern "C" void kernel_launch(void* const* d_in, const int* in_sizes, int n_in,
                              void* d_out, int out_size, void* d_ws, size_t ws_size,
                              hipStream_t stream) {
    const float* points = (const float*)d_in[0];
    const float* feats  = (const float*)d_in[1];
    const float* w1  = (const float*)d_in[2];
    const float* b1  = (const float*)d_in[3];
    const float* g1  = (const float*)d_in[4];
    const float* be1 = (const float*)d_in[5];
    const float* m1  = (const float*)d_in[6];
    const float* v1  = (const float*)d_in[7];
    const float* w2  = (const float*)d_in[8];
    const float* b2  = (const float*)d_in[9];
    const float* cw1 = (const float*)d_in[10];
    const float* cb1 = (const float*)d_in[11];
    const float* cg1 = (const float*)d_in[12];
    const float* cbe1= (const float*)d_in[13];
    const float* cm1 = (const float*)d_in[14];
    const float* cv1 = (const float*)d_in[15];
    const float* cw2 = (const float*)d_in[16];
    const float* cb2 = (const float*)d_in[17];
    const float* cg2 = (const float*)d_in[18];
    const float* cbe2= (const float*)d_in[19];
    const float* cm2 = (const float*)d_in[20];
    const float* cv2 = (const float*)d_in[21];

    char* ws = (char*)d_ws;
    _Float16* w1t  = (_Float16*)(ws + OFF_W1T);
    _Float16* w2t  = (_Float16*)(ws + OFF_W2T);
    _Float16* cw1t = (_Float16*)(ws + OFF_CW1T);
    _Float16* cw2t = (_Float16*)(ws + OFF_CW2T);
    float* b1f  = (float*)(ws + OFF_B1);
    float* b2f  = (float*)(ws + OFF_B2);
    float* cb1f = (float*)(ws + OFF_CB1);
    float* cb2f = (float*)(ws + OFF_CB2);
    float*    bev = (float*)(ws + OFF_BEV);
    _Float16* mid = (_Float16*)(ws + OFF_MID);

    hipMemsetAsync(bev, 0, (size_t)BB * HH * WW * COUT * sizeof(float), stream);

    prep_kernel<<<128, 256, 0, stream>>>(
        w1, b1, g1, be1, m1, v1, w2, b2,
        cw1, cb1, cg1, cbe1, cm1, cv1,
        cw2, cb2, cg2, cbe2, cm2, cv2,
        w1t, w2t, cw1t, cw2t, b1f, b2f, cb1f, cb2f);

    // 400000 pts / 16 per wave / 2 waves per block = 12500 blocks
    mlp_scatter_kernel<<<12500, 64, 0, stream>>>(
        points, feats, w1t, b1f, w2t, b2f, bev);

    // B * H * (W/128) = 4*256*2 = 2048 blocks
    conv3x3_kernel<1><<<2048, 256, 0, stream>>>(bev, cw1t, cb1f, mid);
    conv3x3_kernel<0><<<2048, 256, 0, stream>>>(mid, cw2t, cb2f, d_out);
}
